// PointNet2SetAbstraction_35098472743103
// MI455X (gfx1250) — matrix operand order repaired and checked by reading, not compiled, not run on hardware
//
#include <hip/hip_runtime.h>
#include <hip/hip_bf16.h>

typedef __attribute__((ext_vector_type(16))) _Float16 v16h;
typedef __attribute__((ext_vector_type(8)))  _Float16 v8h;
typedef __attribute__((ext_vector_type(8)))  float    v8f;

#define B_    16
#define N_    4096
#define CPTS  64
#define S_    1024
#define K_    32
#define ROWS  (B_ * S_ * K_)   /* 524288 */
#define IC0   67
#define IC0P  96               /* padded to multiple of 32 for WMMA K loop */

/* ------------------------------------------------------------------ */
/* 1. Farthest point sampling: one workgroup per batch, xyz in LDS.    */
/*    Emits new_xyz directly (fps indices only feed new_xyz).          */
/* ------------------------------------------------------------------ */
__global__ __launch_bounds__(1024)
void fps_kernel(const float* __restrict__ xyz, float* __restrict__ new_xyz) {
    __shared__ float sx[N_], sy[N_], sz[N_];
    __shared__ float rv[32];
    __shared__ int   ri[32];
    __shared__ int   sBest;
    const int b    = blockIdx.x;
    const int tid  = threadIdx.x;
    const int lane = tid & 31;
    const int wave = tid >> 5;
    const float* base = xyz + (size_t)b * N_ * 3;
    for (int p = tid; p < N_; p += 1024) {
        sx[p] = base[p * 3 + 0];
        sy[p] = base[p * 3 + 1];
        sz[p] = base[p * 3 + 2];
    }
    __syncthreads();
    float dmin[4];
#pragma unroll
    for (int t = 0; t < 4; ++t) dmin[t] = 3.4e38f;
    int cur = 0;
    for (int it = 0; it < S_; ++it) {
        const float cx = sx[cur], cy = sy[cur], cz = sz[cur];
        if (tid == 0) {
            float* o = new_xyz + ((size_t)b * S_ + it) * 3;
            o[0] = cx; o[1] = cy; o[2] = cz;
        }
        float bv = -1.0f; int bidx = 0;
#pragma unroll
        for (int t = 0; t < 4; ++t) {
            const int p = tid + t * 1024;
            const float dx = sx[p] - cx, dy = sy[p] - cy, dz = sz[p] - cz;
            const float dd = dx * dx + dy * dy + dz * dz;
            dmin[t] = fminf(dmin[t], dd);
            if (dmin[t] > bv) { bv = dmin[t]; bidx = p; }
        }
#pragma unroll
        for (int off = 16; off > 0; off >>= 1) {
            const float ov = __shfl_down(bv, off, 32);
            const int   oi = __shfl_down(bidx, off, 32);
            if (ov > bv) { bv = ov; bidx = oi; }
        }
        if (lane == 0) { rv[wave] = bv; ri[wave] = bidx; }
        __syncthreads();
        if (wave == 0) {
            bv = rv[lane]; bidx = ri[lane];
#pragma unroll
            for (int off = 16; off > 0; off >>= 1) {
                const float ov = __shfl_down(bv, off, 32);
                const int   oi = __shfl_down(bidx, off, 32);
                if (ov > bv) { bv = ov; bidx = oi; }
            }
            if (lane == 0) sBest = bidx;
        }
        __syncthreads();
        cur = sBest;
    }
}

/* ------------------------------------------------------------------ */
/* 2. KNN: one thread per query, register-resident top-32, LDS tiles.  */
/* ------------------------------------------------------------------ */
#define KNN_TS 1024
__global__ __launch_bounds__(128)
void knn_kernel(const float* __restrict__ xyz, const float* __restrict__ new_xyz,
                int* __restrict__ knn_idx) {
    __shared__ float tx[KNN_TS], ty[KNN_TS], tz[KNN_TS];
    const int b = blockIdx.x >> 3;                 /* 8 blocks/batch */
    const int q = (blockIdx.x & 7) * 128 + threadIdx.x;
    const float* nx = new_xyz + ((size_t)b * S_ + q) * 3;
    const float qx = nx[0], qy = nx[1], qz = nx[2];
    float bd[K_]; int bi[K_];
#pragma unroll
    for (int i = 0; i < K_; ++i) { bd[i] = 3.4e38f; bi[i] = 0; }
    float worst = 3.4e38f;
    const float* base = xyz + (size_t)b * N_ * 3;
    for (int t0 = 0; t0 < N_; t0 += KNN_TS) {
        for (int i = threadIdx.x; i < KNN_TS; i += 128) {
            const float* p = base + (size_t)(t0 + i) * 3;
            tx[i] = p[0]; ty[i] = p[1]; tz[i] = p[2];
        }
        __syncthreads();
        for (int j = 0; j < KNN_TS; ++j) {
            const float dx = tx[j] - qx, dy = ty[j] - qy, dz = tz[j] - qz;
            const float d = dx * dx + dy * dy + dz * dz;
            if (d < worst) {
                bool done = false;
#pragma unroll
                for (int i = 0; i < K_; ++i)
                    if (!done && bd[i] == worst) { bd[i] = d; bi[i] = t0 + j; done = true; }
                worst = bd[0];
#pragma unroll
                for (int i = 1; i < K_; ++i) worst = fmaxf(worst, bd[i]);
            }
        }
        __syncthreads();
    }
    int* o = knn_idx + ((size_t)b * S_ + q) * K_;
#pragma unroll
    for (int i = 0; i < K_; ++i) o[i] = bi[i];
}

/* ------------------------------------------------------------------ */
/* 3. Gather + concat -> padded f16 activations (stride 96).           */
/*    One wave per (b,s,k) row; lanes stride the channels.             */
/* ------------------------------------------------------------------ */
__global__ __launch_bounds__(256)
void gather_kernel(const float* __restrict__ xyz, const float* __restrict__ pts,
                   const float* __restrict__ new_xyz, const int* __restrict__ knn_idx,
                   _Float16* __restrict__ X0) {
    const int row  = blockIdx.x * 8 + (threadIdx.x >> 5);
    const int lane = threadIdx.x & 31;
    const int b = row >> 15;                 /* 32768 rows per batch */
    const int s = (row >> 5) & (S_ - 1);
    const int idx = knn_idx[row];
    const float* pxyz = xyz + ((size_t)b * N_ + idx) * 3;
    const float* pnew = new_xyz + ((size_t)b * S_ + s) * 3;
    const float* pf   = pts + ((size_t)b * N_ + idx) * CPTS;
    _Float16* o = X0 + (size_t)row * IC0P;
#pragma unroll
    for (int t = 0; t < 3; ++t) {
        const int c = lane + t * 32;
        float v;
        if (c < 3)              v = pxyz[c] - pnew[c];
        else if (c < 3 + CPTS)  v = pf[c - 3];
        else                    v = 0.0f;
        o[c] = (_Float16)v;
    }
}

/* ------------------------------------------------------------------ */
/* 4. WMMA GEMM: Y(R x oc) = X(R x icp) * W^T, f16 in / f32 acc.       */
/*    One wave per 16x16 output tile; ISA 7.12.2 wave32 layouts.       */
/* ------------------------------------------------------------------ */
__global__ __launch_bounds__(256)
void gemm_wmma_kernel(const _Float16* __restrict__ X, const _Float16* __restrict__ W,
                      float* __restrict__ Y, int icp, int oc) {
    const int lane = threadIdx.x & 31;
    const int wave = threadIdx.x >> 5;
    const int row0 = (blockIdx.x * 8 + wave) * 16;
    const int oc0  = blockIdx.y * 16;
    const int m  = lane & 15;
    const int hi = lane >> 4;   /* lane group: 0 or 1 */
    v8f acc = {};
    /* A: lane<16 -> row m, K 0..7 (v0-3) & 16..23 (v4-7); lane>=16 -> K 8..15 & 24..31 */
    const _Float16* arow = X + (size_t)(row0 + m) * icp + hi * 8;
    /* B: lane<16 -> col oc0+m, K 0..7 & 16..23; lane>=16 -> K 8..15 & 24..31 (row-major weight row) */
    const _Float16* brow = W + (size_t)(oc0 + m) * icp + hi * 8;
    for (int kb = 0; kb < icp; kb += 32) {
        const v8h a0 = *reinterpret_cast<const v8h*>(arow + kb);
        const v8h a1 = *reinterpret_cast<const v8h*>(arow + kb + 16);
        const v8h b0 = *reinterpret_cast<const v8h*>(brow + kb);
        const v8h b1 = *reinterpret_cast<const v8h*>(brow + kb + 16);
        v16h a, bm;
#pragma unroll
        for (int i = 0; i < 8; ++i) {
            a[i] = a0[i];  a[i + 8] = a1[i];
            bm[i] = b0[i]; bm[i + 8] = b1[i];
        }
        acc = __builtin_amdgcn_wmma_f32_16x16x32_f16(
            false, a, false, bm, (short)0, acc, false, false);
    }
    /* D: vgpr r -> row (r + hi*8), col oc0+m */
    float* yb = Y + (size_t)(row0 + hi * 8) * oc + oc0 + m;
#pragma unroll
    for (int r = 0; r < 8; ++r) yb[(size_t)r * oc] = acc[r];
}

/* ------------------------------------------------------------------ */
/* 5. Helpers: weight pad/convert, stats, finalize, BN+ReLU apply.     */
/* ------------------------------------------------------------------ */
__global__ void wpad_kernel(const float* __restrict__ W, _Float16* __restrict__ Wp,
                            int oc, int ic, int icp) {
    const int e = blockIdx.x * 256 + threadIdx.x;
    if (e >= oc * icp) return;
    const int r = e / icp, c = e % icp;
    Wp[e] = (_Float16)(c < ic ? W[r * ic + c] : 0.0f);
}

__global__ void zstats_kernel(float* st) {
    st[threadIdx.x] = 0.0f;   /* gsum[128] + gsq[128] */
}

__global__ __launch_bounds__(256)
void stats_kernel(const float* __restrict__ Y, float* __restrict__ gsum,
                  float* __restrict__ gsq, int oc, size_t total) {
    __shared__ float s1[128], s2[128];
    for (int c = threadIdx.x; c < oc; c += 256) { s1[c] = 0.0f; s2[c] = 0.0f; }
    __syncthreads();
    const size_t stride = (size_t)gridDim.x * 256;
    for (size_t e = (size_t)blockIdx.x * 256 + threadIdx.x; e < total; e += stride) {
        const float v = Y[e];
        const int c = (int)(e & (size_t)(oc - 1));
        atomicAdd(&s1[c], v);
        atomicAdd(&s2[c], v * v);
    }
    __syncthreads();
    for (int c = threadIdx.x; c < oc; c += 256) {
        atomicAdd(&gsum[c], s1[c]);
        atomicAdd(&gsq[c],  s2[c]);
    }
}

__global__ void finalize_kernel(const float* __restrict__ gsum, const float* __restrict__ gsq,
                                const float* __restrict__ gamma, const float* __restrict__ beta,
                                float* __restrict__ A, float* __restrict__ Bv, int oc) {
    const int c = threadIdx.x;
    if (c >= oc) return;
    const float inv = 1.0f / (float)ROWS;
    const float mu  = gsum[c] * inv;
    const float var = gsq[c] * inv - mu * mu;
    const float aa  = gamma[c] * rsqrtf(var + 1e-5f);
    A[c]  = aa;
    Bv[c] = beta[c] - mu * aa;
}

__global__ __launch_bounds__(256)
void apply_kernel(const float* __restrict__ Y, const float* __restrict__ A,
                  const float* __restrict__ Bv, _Float16* __restrict__ Xn,
                  int oc, size_t total) {
    const size_t stride = (size_t)gridDim.x * 256;
    for (size_t e = (size_t)blockIdx.x * 256 + threadIdx.x; e < total; e += stride) {
        const int c = (int)(e & (size_t)(oc - 1));
        const float v = Y[e] * A[c] + Bv[c];
        Xn[e] = (_Float16)fmaxf(v, 0.0f);
    }
}

/* ------------------------------------------------------------------ */
/* 6. Fused BN + ReLU + max over K=32 samples for the last layer.      */
/* ------------------------------------------------------------------ */
__global__ __launch_bounds__(128)
void maxpool_kernel(const float* __restrict__ Y, const float* __restrict__ A,
                    const float* __restrict__ Bv, float* __restrict__ out) {
    const int p = blockIdx.x;   /* b*S + s */
    const int c = threadIdx.x;  /* 0..127  */
    const float aa = A[c], bb = Bv[c];
    const float* yb = Y + (size_t)p * K_ * 128 + c;
    float m = 0.0f;             /* relu output >= 0 */
#pragma unroll
    for (int k = 0; k < K_; ++k) {
        const float v = yb[(size_t)k * 128] * aa + bb;
        m = fmaxf(m, fmaxf(v, 0.0f));
    }
    out[(size_t)p * 128 + c] = m;
}

/* ------------------------------------------------------------------ */
extern "C" void kernel_launch(void* const* d_in, const int* in_sizes, int n_in,
                              void* d_out, int out_size, void* d_ws, size_t ws_size,
                              hipStream_t stream) {
    (void)in_sizes; (void)n_in; (void)out_size; (void)ws_size;
    const float* xyz = (const float*)d_in[0];
    const float* pts = (const float*)d_in[1];
    const float* W0  = (const float*)d_in[2];
    const float* g0  = (const float*)d_in[3];
    const float* be0 = (const float*)d_in[4];
    const float* W1  = (const float*)d_in[5];
    const float* g1  = (const float*)d_in[6];
    const float* be1 = (const float*)d_in[7];
    const float* W2  = (const float*)d_in[8];
    const float* g2  = (const float*)d_in[9];
    const float* be2 = (const float*)d_in[10];

    float* out      = (float*)d_out;
    float* new_xyz  = out;                          /* B*S*3 = 49152 */
    float* xout     = out + (size_t)B_ * S_ * 3;    /* B*S*128       */

    char* ws = (char*)d_ws;
    int*      knn  = (int*)(ws + 0);                /* 2,097,152 B */
    float*    st   = (float*)(ws + 2097152);        /* 2048 B      */
    float*    gsum = st, *gsq = st + 128, *Aa = st + 256, *Bb = st + 384;
    _Float16* W0p  = (_Float16*)(ws + 2099200);     /* 64*96  f16 */
    _Float16* W1p  = (_Float16*)(ws + 2111488);     /* 64*64  f16 */
    _Float16* W2p  = (_Float16*)(ws + 2119680);     /* 128*64 f16 */
    _Float16* X    = (_Float16*)(ws + 2136064);     /* ROWS*96 f16 (reused stride 64) */
    float*    Yb   = (float*)(ws + 102799360);      /* ROWS*128 f32 */

    /* geometry */
    fps_kernel<<<B_, 1024, 0, stream>>>(xyz, new_xyz);
    knn_kernel<<<B_ * (S_ / 128), 128, 0, stream>>>(xyz, new_xyz, knn);

    /* weights -> padded f16 */
    wpad_kernel<<<(64 * 96  + 255) / 256, 256, 0, stream>>>(W0, W0p, 64, IC0, IC0P);
    wpad_kernel<<<(64 * 64  + 255) / 256, 256, 0, stream>>>(W1, W1p, 64, 64, 64);
    wpad_kernel<<<(128 * 64 + 255) / 256, 256, 0, stream>>>(W2, W2p, 128, 64, 64);

    gather_kernel<<<ROWS / 8, 256, 0, stream>>>(xyz, pts, new_xyz, knn, X);

    const size_t n64  = (size_t)ROWS * 64;
    const size_t n128 = (size_t)ROWS * 128;

    /* layer 0: icp=96 -> oc=64 */
    zstats_kernel<<<1, 256, 0, stream>>>(st);
    gemm_wmma_kernel<<<dim3(ROWS / 128, 4), 256, 0, stream>>>(X, W0p, Yb, IC0P, 64);
    stats_kernel<<<1024, 256, 0, stream>>>(Yb, gsum, gsq, 64, n64);
    finalize_kernel<<<1, 128, 0, stream>>>(gsum, gsq, g0, be0, Aa, Bb, 64);
    apply_kernel<<<4096, 256, 0, stream>>>(Yb, Aa, Bb, X, 64, n64);

    /* layer 1: 64 -> 64 */
    zstats_kernel<<<1, 256, 0, stream>>>(st);
    gemm_wmma_kernel<<<dim3(ROWS / 128, 4), 256, 0, stream>>>(X, W1p, Yb, 64, 64);
    stats_kernel<<<1024, 256, 0, stream>>>(Yb, gsum, gsq, 64, n64);
    finalize_kernel<<<1, 128, 0, stream>>>(gsum, gsq, g1, be1, Aa, Bb, 64);
    apply_kernel<<<4096, 256, 0, stream>>>(Yb, Aa, Bb, X, 64, n64);

    /* layer 2: 64 -> 128, fused BN+ReLU+maxpool */
    zstats_kernel<<<1, 256, 0, stream>>>(st);
    gemm_wmma_kernel<<<dim3(ROWS / 128, 8), 256, 0, stream>>>(X, W2p, Yb, 64, 128);
    stats_kernel<<<1024, 256, 0, stream>>>(Yb, gsum, gsq, 128, n128);
    finalize_kernel<<<1, 128, 0, stream>>>(gsum, gsq, g2, be2, Aa, Bb, 128);
    maxpool_kernel<<<B_ * S_, 128, 0, stream>>>(Yb, Aa, Bb, xout);
}
